// OnlineTripletLoss_53360673686073
// MI455X (gfx1250) — compile-verified, matrix operand order checked by reading, not run on hardware
//
#include <hip/hip_runtime.h>
#include <hip/hip_bf16.h>

#define B_ROWS 8192
#define DIMS 256
#define MARGIN 0.5f
#define NSLICE 4
#define COLS_PER_SLICE (B_ROWS / NSLICE)   // 2048
#define MT 128                              // rows per workgroup
#define NT 128                              // cols per tile
#define KT 32                               // K per WMMA step

typedef __attribute__((ext_vector_type(16))) __bf16 v16bf;
typedef __attribute__((ext_vector_type(8)))  float  v8f;
typedef __attribute__((ext_vector_type(4)))  unsigned int v4u;
typedef __attribute__((ext_vector_type(8)))  unsigned int v8u;

union FragB {
    v16bf v;
    uint4 q[2];
};

// ---------------------------------------------------------------------------
// TDM: issue a 2D tile load (NT rows x KT bf16, row stride DIMS) into LDS.
// D# group0/group1 per CDNA5 ISA; 2-group form (2D tensor,
// VADDR2/VADDR3 disabled). Tracked by TENSORcnt.
// ---------------------------------------------------------------------------
__device__ __forceinline__ void tdm_load_tile_bf16(const unsigned short* tile,
                                                   unsigned ldsOff) {
    unsigned long long ga = (unsigned long long)(size_t)tile;
    v4u g0;
    g0[0] = 1u;                                            // count=1 (valid)
    g0[1] = ldsOff;                                        // lds_addr
    g0[2] = (unsigned)ga;                                  // global_addr[31:0]
    g0[3] = ((unsigned)(ga >> 32) & 0x01FFFFFFu)           // global_addr[56:32]
            | (2u << 30);                                  // type=2 ("image")
    v8u g1;
    g1[0] = (1u << 16);                                    // data_size=1 -> 2B
    g1[1] = ((unsigned)DIMS   & 0xFFFFu) << 16;            // tensor_dim0=256
    g1[2] = ((unsigned)B_ROWS & 0xFFFFu) << 16;            // tensor_dim1=8192
    g1[3] = ((unsigned)KT) << 16;                          // tile_dim0=32
    g1[4] = (unsigned)NT;                                  // tile_dim1=128
    g1[5] = (unsigned)DIMS;                                // dim0_stride=256
    g1[6] = 0u;
    g1[7] = 0u;
    asm volatile("tensor_load_to_lds %0, %1"
                 :
                 : "s"(g0), "s"(g1)
                 : "memory");
}

// ---------------------------------------------------------------------------
// Kernel 1: split f32 -> bf16 hi/lo pair, compute per-row sum of squares.
// ---------------------------------------------------------------------------
__global__ void prep_kernel(const float* __restrict__ emb,
                            unsigned short* __restrict__ hi,
                            unsigned short* __restrict__ lo,
                            float* __restrict__ sq) {
    const int wave = threadIdx.x >> 5;
    const int lane = threadIdx.x & 31;
    const int row  = blockIdx.x * 8 + wave;

    const float* rp = emb + row * DIMS + lane * 8;
    float4 a = ((const float4*)rp)[0];
    float4 b = ((const float4*)rp)[1];
    float v[8] = {a.x, a.y, a.z, a.w, b.x, b.y, b.z, b.w};

    union { unsigned short u[8]; uint4 q; } H, L;
    float ss = 0.0f;
#pragma unroll
    for (int i = 0; i < 8; ++i) {
        float x = v[i];
        ss += x * x;
        __hip_bfloat16 h = __float2bfloat16(x);         // RNE
        float hf = __bfloat162float(h);
        __hip_bfloat16 l = __float2bfloat16(x - hf);    // residual
        H.u[i] = *(unsigned short*)&h;
        L.u[i] = *(unsigned short*)&l;
    }
    *(uint4*)(hi + row * DIMS + lane * 8) = H.q;
    *(uint4*)(lo + row * DIMS + lane * 8) = L.q;

#pragma unroll
    for (int m = 1; m < 32; m <<= 1) ss += __shfl_xor(ss, m, 32);
    if (lane == 0) sq[row] = ss;
}

// ---------------------------------------------------------------------------
// Kernel 2: fused Gram (bf16x3 WMMA) + semi-hard mining epilogue.
// grid = (64 row-blocks, 4 column slices), 256 threads = 8 wave32.
// A fragments resident in VGPRs; B k-slices delivered by TDM into
// double-buffered LDS, overlapped with WMMA via TENSORcnt.
// ---------------------------------------------------------------------------
__global__ void __launch_bounds__(256)
gram_mine_kernel(const unsigned short* __restrict__ hi,
                 const unsigned short* __restrict__ lo,
                 const float* __restrict__ sq,
                 const int* __restrict__ labels,
                 float* __restrict__ outMaxPos,   // [NSLICE][B_ROWS]
                 float* __restrict__ outMinNeg) { // [NSLICE][B_ROWS]
    __shared__ unsigned short sHi[2][NT * KT];   // 2 x 8 KB
    __shared__ unsigned short sLo[2][NT * KT];   // 2 x 8 KB
    __shared__ int   sLab[NT];
    __shared__ float sSq[NT];

    const int tid  = threadIdx.x;
    const int wave = tid >> 5;
    const int lane = tid & 31;
    const int half = lane >> 4;      // K-half / M-half selector
    const int l15  = lane & 15;
    const int rowBlock = blockIdx.x;
    const int slice    = blockIdx.y;
    const int waveRow0 = rowBlock * MT + wave * 16;

    const unsigned ldsHiOff[2] = { (unsigned)(size_t)&sHi[0][0],
                                   (unsigned)(size_t)&sHi[1][0] };
    const unsigned ldsLoOff[2] = { (unsigned)(size_t)&sLo[0][0],
                                   (unsigned)(size_t)&sLo[1][0] };

    // --- persistent A fragments: 16 rows x 256 K, hi and lo (128 VGPRs) ----
    v16bf aHi[8], aLo[8];
    {
        const int arow = waveRow0 + l15;
        const uint4* gh = (const uint4*)(hi + arow * DIMS);
        const uint4* gl = (const uint4*)(lo + arow * DIMS);
#pragma unroll
        for (int ks = 0; ks < 8; ++ks) {
            int base = (ks * 32 + half * 16) >> 3;   // uint4 == 8 bf16
            uint4* qh = (uint4*)&aHi[ks];
            uint4* ql = (uint4*)&aLo[ks];
            qh[0] = gh[base]; qh[1] = gh[base + 1];
            ql[0] = gl[base]; ql[1] = gl[base + 1];
        }
    }

    // per-row metadata (row = waveRow0 + r + 8*half, per C/D VGPR layout)
    int   rLab[8];
    float rSq[8];
#pragma unroll
    for (int r = 0; r < 8; ++r) {
        int rr = waveRow0 + r + 8 * half;
        rLab[r] = labels[rr];
        rSq[r]  = sq[rr];
    }

    float maxp[8], minn[8];
#pragma unroll
    for (int r = 0; r < 8; ++r) { maxp[r] = -1.0f; minn[r] = 1e30f; }

    const int colSliceBase = slice * COLS_PER_SLICE;
    for (int ct = 0; ct < COLS_PER_SLICE / NT; ++ct) {
        const int colBase = colSliceBase + ct * NT;

        __syncthreads();                       // prev ct fully consumed
        if (tid < NT) {
            sLab[tid] = labels[colBase + tid];
            sSq[tid]  = sq[colBase + tid];
        }
        if (tid < 32) {                        // wave 0: prime stage 0 -> buf0
            const unsigned short* tb = hi + (size_t)colBase * DIMS;
            tdm_load_tile_bf16(tb, ldsHiOff[0]);
            const unsigned short* tl = lo + (size_t)colBase * DIMS;
            tdm_load_tile_bf16(tl, ldsLoOff[0]);
        }

        v8f acc[8] = {};                       // 8 tiles of 16x16 f32

        for (int ks = 0; ks < 8; ++ks) {
            const int cur = ks & 1;
            __syncthreads();                   // buffer cur^1 free to overwrite
            if (tid < 32) {
                if (ks + 1 < 8) {              // prefetch next stage -> cur^1
                    const size_t koff = (size_t)colBase * DIMS + (ks + 1) * KT;
                    tdm_load_tile_bf16(hi + koff, ldsHiOff[cur ^ 1]);
                    tdm_load_tile_bf16(lo + koff, ldsLoOff[cur ^ 1]);
                    __builtin_amdgcn_s_wait_tensorcnt((short)2); // cur landed
                } else {
                    __builtin_amdgcn_s_wait_tensorcnt((short)0); // all landed
                }
            }
            __syncthreads();                   // cur data visible to all waves

#pragma unroll
            for (int t = 0; t < 8; ++t) {
                FragB bh, bl;
                int nl = t * 16 + l15;
                const uint4* ph = (const uint4*)&sHi[cur][nl * KT + half * 16];
                const uint4* pl = (const uint4*)&sLo[cur][nl * KT + half * 16];
                bh.q[0] = ph[0]; bh.q[1] = ph[1];
                bl.q[0] = pl[0]; bl.q[1] = pl[1];
                // bf16x3 split: hi*hi + hi*lo + lo*hi -> ~fp32 dot product
                acc[t] = __builtin_amdgcn_wmma_f32_16x16x32_bf16(
                    false, aHi[ks], false, bh.v, (short)0, acc[t], false, false);
                acc[t] = __builtin_amdgcn_wmma_f32_16x16x32_bf16(
                    false, aHi[ks], false, bl.v, (short)0, acc[t], false, false);
                acc[t] = __builtin_amdgcn_wmma_f32_16x16x32_bf16(
                    false, aLo[ks], false, bh.v, (short)0, acc[t], false, false);
            }
        }

        // ---- fused mining epilogue for this 128-column tile ----
#pragma unroll
        for (int t = 0; t < 8; ++t) {
            int   col  = colBase + t * 16 + l15;
            int   clab = sLab[t * 16 + l15];
            float csq  = sSq[t * 16 + l15];
#pragma unroll
            for (int r = 0; r < 8; ++r) {
                int   row  = waveRow0 + r + 8 * half;
                float dot  = acc[t][r];
                float d2   = fmaxf(rSq[r] + csq - 2.0f * dot, 0.0f);
                float dist = sqrtf(d2);
                bool  same = (clab == rLab[r]);
                if (same && (row != col)) maxp[r] = fmaxf(maxp[r], dist);
                if (!same)                minn[r] = fminf(minn[r], dist);
            }
        }
    }

    // reduce across the 16 lanes of each half (wave32: masks 1..8 stay in half)
#pragma unroll
    for (int r = 0; r < 8; ++r) {
        float mp = maxp[r], mn = minn[r];
#pragma unroll
        for (int m = 1; m < 16; m <<= 1) {
            mp = fmaxf(mp, __shfl_xor(mp, m, 32));
            mn = fminf(mn, __shfl_xor(mn, m, 32));
        }
        if (l15 == 0) {
            int row = waveRow0 + r + 8 * half;
            outMaxPos[slice * B_ROWS + row] = mp;
            outMinNeg[slice * B_ROWS + row] = mn;
        }
    }
}

// ---------------------------------------------------------------------------
// Kernel 3: merge column slices, apply margin & validity per row.
// ---------------------------------------------------------------------------
__global__ void combine_kernel(const float* __restrict__ mp,
                               const float* __restrict__ mn,
                               float* __restrict__ rowLoss,
                               float* __restrict__ rowCnt) {
    int i = blockIdx.x * blockDim.x + threadIdx.x;
    float dap = -1.0f, dan = 1e30f;
#pragma unroll
    for (int s = 0; s < NSLICE; ++s) {
        dap = fmaxf(dap, mp[s * B_ROWS + i]);
        dan = fminf(dan, mn[s * B_ROWS + i]);
    }
    bool has_pos = (dap >= 0.0f);     // distances are >= 0; init was -1
    bool has_neg = (dan < 1e29f);
    bool valid   = has_pos && has_neg && (dan < dap + MARGIN);
    float loss   = fmaxf(dap - dan + MARGIN, 0.0f);
    rowLoss[i] = valid ? loss : 0.0f;
    rowCnt[i]  = valid ? 1.0f : 0.0f;
}

// ---------------------------------------------------------------------------
// Kernel 4: deterministic single-block reduction -> scalar mean.
// ---------------------------------------------------------------------------
__global__ void reduce_kernel(const float* __restrict__ rowLoss,
                              const float* __restrict__ rowCnt,
                              float* __restrict__ out) {
    __shared__ float sL[256], sC[256];
    int t = threadIdx.x;
    float l = 0.0f, c = 0.0f;
    for (int i = t; i < B_ROWS; i += 256) { l += rowLoss[i]; c += rowCnt[i]; }
    sL[t] = l; sC[t] = c;
    __syncthreads();
    for (int s = 128; s > 0; s >>= 1) {
        if (t < s) { sL[t] += sL[t + s]; sC[t] += sC[t + s]; }
        __syncthreads();
    }
    if (t == 0) out[0] = (sC[0] > 0.0f) ? (sL[0] / sC[0]) : 0.0f;
}

// ---------------------------------------------------------------------------
extern "C" void kernel_launch(void* const* d_in, const int* in_sizes, int n_in,
                              void* d_out, int out_size, void* d_ws, size_t ws_size,
                              hipStream_t stream) {
    (void)in_sizes; (void)n_in; (void)out_size; (void)ws_size;
    const float* emb    = (const float*)d_in[0];
    const int*   labels = (const int*)d_in[1];
    float*       out    = (float*)d_out;

    char* ws = (char*)d_ws;
    unsigned short* hi = (unsigned short*)ws;  ws += (size_t)B_ROWS * DIMS * 2;
    unsigned short* lo = (unsigned short*)ws;  ws += (size_t)B_ROWS * DIMS * 2;
    float* sq      = (float*)ws;               ws += (size_t)B_ROWS * 4;
    float* mp      = (float*)ws;               ws += (size_t)NSLICE * B_ROWS * 4;
    float* mn      = (float*)ws;               ws += (size_t)NSLICE * B_ROWS * 4;
    float* rowLoss = (float*)ws;               ws += (size_t)B_ROWS * 4;
    float* rowCnt  = (float*)ws;

    prep_kernel<<<B_ROWS / 8, 256, 0, stream>>>(emb, hi, lo, sq);

    dim3 grid(B_ROWS / MT, NSLICE);
    gram_mine_kernel<<<grid, 256, 0, stream>>>(hi, lo, sq, labels, mp, mn);

    combine_kernel<<<B_ROWS / 256, 256, 0, stream>>>(mp, mn, rowLoss, rowCnt);
    reduce_kernel<<<1, 256, 0, stream>>>(rowLoss, rowCnt, out);
}
